// ResidualVectorQuantizer_89086211653873
// MI455X (gfx1250) — matrix operand order repaired and checked by reading, not compiled, not run on hardware
//
#include <hip/hip_runtime.h>
#include <math.h>

typedef __attribute__((ext_vector_type(16))) __bf16 v16bf;
typedef __attribute__((ext_vector_type(8)))  float  v8f;

#define B_   32
#define CD   256      // C == D
#define HW   4096     // 64*64
#define NROW (B_*HW)  // 131072
#define LV   4
#define KC_  1024
#define RPB  128      // rows per block
#define NBLK (NROW/RPB)  // 1024
#define NTHR 256
#define BETA_ 0.25f

// round-to-nearest-even float -> bf16 (bit manipulation; no __bf16 arithmetic)
static __device__ __forceinline__ __bf16 f2bf(float f) {
  unsigned u = __builtin_bit_cast(unsigned, f);
  unsigned r = u + 0x7FFFu + ((u >> 16) & 1u);
  unsigned short h = (unsigned short)(r >> 16);
  return __builtin_bit_cast(__bf16, h);
}

static __device__ __forceinline__ void wait_async0() {
#if __has_builtin(__builtin_amdgcn_s_wait_asynccnt)
  __builtin_amdgcn_s_wait_asynccnt(0);
#else
  asm volatile("s_wait_asynccnt 0" ::: "memory");
#endif
}

// 32 B per lane: memory[saddr + goff (+0/16)] -> LDS[ldsAddr (+0/16)], ASYNCcnt-tracked.
// Per ISA §10 async pseudocode, INST_OFFSET applies to both the LDS and memory address.
static __device__ __forceinline__ void async_stage32(unsigned ldsAddr, unsigned goff,
                                                     const void* sbase) {
  asm volatile("global_load_async_to_lds_b128 %0, %1, %2\n\t"
               "global_load_async_to_lds_b128 %0, %1, %2 offset:16"
               :: "v"(ldsAddr), "v"(goff), "s"(sbase)
               : "memory");
}

// Pre-pack emb^T (per level) into WMMA B-fragment order, bf16.
// Fragment (l, kc, dc): 32(d) x 16(k); col=lane%16, h=lane/16,
// elem i -> d = d0 + 8h + (i&7) + (i>>3)*16.
// Layout: embB[((l*64+kc)*8+dc)*512 + lane*16 + i]
__global__ void rvq_pack_embB(const float* __restrict__ embs, __bf16* __restrict__ embB) {
  unsigned flat = blockIdx.x * blockDim.x + threadIdx.x;  // < 1,048,576
  unsigned i    = flat & 15u;
  unsigned lane = (flat >> 4) & 31u;
  unsigned dc   = (flat >> 9) & 7u;
  unsigned kc   = (flat >> 12) & 63u;
  unsigned l    = flat >> 18;
  unsigned k = kc * 16u + (lane & 15u);
  unsigned d = dc * 32u + (lane >> 4) * 8u + (i & 7u) + (i >> 3) * 16u;
  embB[flat] = f2bf(embs[((size_t)l * KC_ + k) * CD + d]);
}

// e2[l*K+k] = ||emb||^2 ; also zero histogram (must re-zero every launch).
__global__ void rvq_e2_init(const float* __restrict__ embs, float* __restrict__ e2,
                            unsigned* __restrict__ hist) {
  int t = blockIdx.x * blockDim.x + threadIdx.x;  // < 4096
  const float* row = embs + (size_t)t * CD;
  float s = 0.f;
  for (int d = 0; d < CD; ++d) { float v = row[d]; s += v * v; }
  e2[t] = s;
  hist[t] = 0u;
}

__global__ __launch_bounds__(NTHR) void rvq_main(
    const float* __restrict__ zin, const float* __restrict__ embs,
    const __bf16* __restrict__ embB, const float* __restrict__ e2g,
    unsigned* __restrict__ hist, float* __restrict__ partial,
    float* __restrict__ zq, float* __restrict__ codes)
{
  extern __shared__ char smem[];
  float*  resid  = (float*)smem;                 // [d][row] 128*256 f32 = 128 KB
  __bf16* stage  = (__bf16*)(smem + 131072);     // 2 x 4096 bf16 = 16 KB
  float*  e2s    = (float*)(smem + 147456);      // 1024 f32 = 4 KB
  int*    idxArr = (int*)(smem + 151552);        // 128 ints
  float*  red    = (float*)(smem + 152064);      // 256 f32

  const int tid  = threadIdx.x;
  const int lane = tid & 31;
  const int wave = tid >> 5;
  const int lr   = lane & 15;
  const int half = lane >> 4;
  const int rw0  = wave * 16;       // each wave owns 16 rows

  // LDS byte address of the stage area (ISA: generic->LDS maps via addr[31:0]).
  const unsigned stageLds = (unsigned)(uintptr_t)(void*)stage;

  const int bimg = blockIdx.x >> 5;           // 32 blocks per image
  const int r0   = (blockIdx.x & 31) * RPB;
  const size_t zbase = (size_t)bimg * CD * HW + r0;

  // Load z tile -> LDS residual, [d][row]; rows contiguous in memory per d.
  for (int e = tid; e < RPB * CD; e += NTHR) {
    int row = e & 127, d = e >> 7;
    resid[d * RPB + row] = zin[zbase + (size_t)d * HW + row];
  }
  float lacc = 0.f;

  for (int lvl = 0; lvl < LV; ++lvl) {
    __syncthreads();  // residual tile ready (initial load / previous update)

    // Stage ||e||^2 for this level into LDS.
    for (int t = tid; t < KC_; t += NTHR) e2s[t] = e2g[lvl * KC_ + t];

    // Pack A fragments (16 rows x 32 d, bf16; fold the 2x here).
    v16bf aF[8];
    #pragma unroll
    for (int dc = 0; dc < 8; ++dc) {
      #pragma unroll
      for (int i = 0; i < 16; ++i) {
        int d = dc * 32 + half * 8 + (i & 7) + ((i >> 3) * 16);
        aF[dc][i] = f2bf(2.0f * resid[d * RPB + rw0 + lr]);
      }
    }

    float bestV[8]; int bestK[8];
    #pragma unroll
    for (int j = 0; j < 8; ++j) { bestV[j] = -3.402823466e38f; bestK[j] = 0; }

    // Async-prefetch k-chunk 0 straight into stage buffer 0 (32 B per lane).
    async_stage32(stageLds + tid * 32, (unsigned)(tid * 32),
                  (const void*)(embB + (size_t)(lvl * 64) * 4096));
    wait_async0();
    __syncthreads();

    for (int kc = 0; kc < 64; ++kc) {
      // Prefetch next chunk into the other buffer; its last readers finished
      // before the barrier that ended iteration kc-1, so no WAR hazard.
      if (kc + 1 < 64)
        async_stage32(stageLds + (((kc + 1) & 1) * 8192) + tid * 32,
                      (unsigned)(tid * 32),
                      (const void*)(embB + (size_t)(lvl * 64 + kc + 1) * 4096));

      const __bf16* sb = stage + (kc & 1) * 4096;
      // Two independent 4-deep accumulation chains for matrix-pipe ILP.
      v8f acc0, acc1;
      #pragma unroll
      for (int j = 0; j < 8; ++j) { acc0[j] = 0.f; acc1[j] = 0.f; }
      #pragma unroll
      for (int dc = 0; dc < 4; ++dc) {
        v16bf bF = *(const v16bf*)(sb + dc * 512 + lane * 16);
        acc0 = __builtin_amdgcn_wmma_f32_16x16x32_bf16(false, aF[dc], false, bF,
                                                       (short)0, acc0, false, false);
      }
      #pragma unroll
      for (int dc = 4; dc < 8; ++dc) {
        v16bf bF = *(const v16bf*)(sb + dc * 512 + lane * 16);
        acc1 = __builtin_amdgcn_wmma_f32_16x16x32_bf16(false, aF[dc], false, bF,
                                                       (short)0, acc1, false, false);
      }
      // acc[j]: row = rw0 + half*8 + j, col k = kc*16 + lr
      const int k = kc * 16 + lr;
      const float e2v = e2s[k];
      #pragma unroll
      for (int j = 0; j < 8; ++j) {
        float lg = (acc0[j] + acc1[j]) - e2v;       // 2*r.e - ||e||^2
        if (lg > bestV[j]) { bestV[j] = lg; bestK[j] = k; }  // strict > keeps lowest k
      }
      wait_async0();    // our prefetch landed in LDS
      __syncthreads();  // all waves' prefetches + reads complete
    }

    // Argmax across the 16 columns held by lanes of each half (first-max tiebreak).
    #pragma unroll
    for (int off = 1; off < 16; off <<= 1) {
      #pragma unroll
      for (int j = 0; j < 8; ++j) {
        float ov = __shfl_xor(bestV[j], off, 32);
        int   ok = __shfl_xor(bestK[j], off, 32);
        if (ov > bestV[j] || (ov == bestV[j] && ok < bestK[j])) {
          bestV[j] = ov; bestK[j] = ok;
        }
      }
    }
    if (lr == 0) {
      #pragma unroll
      for (int j = 0; j < 8; ++j) idxArr[rw0 + half * 8 + j] = bestK[j];
    }
    __syncthreads();

    // Codes (as float, (B,L,H,W)) + deterministic integer histogram.
    if (tid < RPB) {
      int k = idxArr[tid];
      codes[((size_t)(bimg * LV + lvl)) * HW + r0 + tid] = (float)k;
      atomicAdd(&hist[lvl * KC_ + k], 1u);
    }

    // residual -= q ; accumulate sum((r_new)^2) == (codebook+commitment)/(1+beta) term
    const float* embL = embs + (size_t)lvl * KC_ * CD;
    for (int e = tid; e < RPB * CD; e += NTHR) {
      int row = e & 127, d = e >> 7;
      float q = embL[(size_t)idxArr[row] * CD + d];
      float r = resid[d * RPB + row] - q;
      resid[d * RPB + row] = r;
      lacc += r * r;
    }
  }
  __syncthreads();

  // z_q = sum_q = x - residual_final (straight-through forward value)
  for (int e = tid; e < RPB * CD; e += NTHR) {
    int row = e & 127, d = e >> 7;
    size_t g = zbase + (size_t)d * HW + row;
    zq[g] = zin[g] - resid[d * RPB + row];
  }

  // Deterministic block loss reduction.
  red[tid] = lacc;
  __syncthreads();
  if (tid == 0) {
    float s = 0.f;
    for (int i = 0; i < NTHR; ++i) s += red[i];
    partial[blockIdx.x] = s;
  }
}

__global__ void rvq_final(const float* __restrict__ partial,
                          const unsigned* __restrict__ hist,
                          float* __restrict__ out) {
  __shared__ float perpL[LV];
  int tid = threadIdx.x;
  if (tid < LV) {
    const unsigned* h = hist + tid * KC_;
    float tot = 0.f;
    for (int k = 0; k < KC_; ++k) tot += (float)h[k];
    float inv = 1.0f / (tot + 1e-9f);
    float ent = 0.f;
    for (int k = 0; k < KC_; ++k) {
      float p = (float)h[k] * inv;
      ent -= p * logf(p + 1e-9f);
    }
    perpL[tid] = expf(ent);
  }
  __syncthreads();
  if (tid == 0) {
    float s = 0.f;
    for (int i = 0; i < NBLK; ++i) s += partial[i];   // fixed order: deterministic
    out[0] = (1.0f + BETA_) * s / (float)((size_t)NROW * CD);  // vq_loss
    float ps = 0.f;
    for (int l = 0; l < LV; ++l) ps += perpL[l];
    out[1] = ps * (1.0f / LV);                                  // perplexity
  }
}

extern "C" void kernel_launch(void* const* d_in, const int* in_sizes, int n_in,
                              void* d_out, int out_size, void* d_ws, size_t ws_size,
                              hipStream_t stream) {
  (void)in_sizes; (void)n_in; (void)out_size; (void)ws_size;
  const float* z    = (const float*)d_in[0];   // (32,256,64,64) f32
  const float* embs = (const float*)d_in[1];   // (4,1024,256) f32

  float* out   = (float*)d_out;
  float* zq    = out;                               // 33,554,432
  float* codes = out + (size_t)B_ * CD * HW;        // + 524,288
  float* scal  = codes + (size_t)B_ * LV * HW;      // loss, perplexity

  char* ws = (char*)d_ws;
  __bf16*   embB    = (__bf16*)ws;                              // 2 MB
  float*    e2      = (float*)(ws + 2097152);                   // 16 KB
  unsigned* hist    = (unsigned*)(ws + 2097152 + 16384);        // 16 KB
  float*    partial = (float*)(ws + 2097152 + 16384 + 16384);   // 4 KB

  rvq_pack_embB<<<4096, 256, 0, stream>>>(embs, embB);
  rvq_e2_init<<<16, 256, 0, stream>>>(embs, e2, hist);
  const size_t smem = 153088;  // 128K resid + 16K stage + 4K e2 + idx + red
  rvq_main<<<NBLK, NTHR, smem, stream>>>(z, embs, embB, e2, hist, partial, zq, codes);
  rvq_final<<<1, 64, 0, stream>>>(partial, hist, scal);
}
